// SparseConvBlock_7035156431607
// MI455X (gfx1250) — matrix operand order repaired and checked by reading, not compiled, not run on hardware
//
#include <hip/hip_runtime.h>

// ---------------------------------------------------------------------------
// SubMConv3d (rulebook gather -> bf16 WMMA GEMM) + BatchNorm + ReLU, gfx1250
//  - B matrix staged to LDS via GLOBAL_LOAD_ASYNC_TO_LDS_B128 (ASYNCcnt)
//  - A gathers software-pipelined (idx prefetched one offset ahead)
// ---------------------------------------------------------------------------

typedef __attribute__((ext_vector_type(16))) __bf16 v16bf;
typedef __attribute__((ext_vector_type(8)))  float  v8f;

#define NV     150000
#define CIN    32
#define COUT   64
#define KOFF   27
#define MT     16                  // output rows per wave tile
#define NTILES (NV / MT)           // 9375 (exact)
#define WPB    8                   // waves per block (256 threads)

#define WPK_ELEMS (KOFF * 4 * 32 * 16)   // 55296 bf16
#define WPK_BYTES (WPK_ELEMS * 2)        // 110592 B (fits in 320KB WGP LDS)

// workspace layout (bytes)
#define FB_OFF    0u               // bf16 features  [NV][CIN]      = 9,600,000 B
#define WPK_OFF   9600000u         // bf16 packed W  [27][4][32][16]=   110,592 B
#define SUM_OFF   9710592u         // f32 sum[64]
#define SQ_OFF    (SUM_OFF + 256u) // f32 sumsq[64]
#define SCALE_OFF (SUM_OFF + 512u) // f32 scale[64]
#define SHIFT_OFF (SUM_OFF + 768u) // f32 shift[64]

union Frag16 { v16bf v; uint4 q[2]; };

// --- 1. features f32 -> bf16 (row major) -----------------------------------
__global__ __launch_bounds__(256) void cvt_feats(const float* __restrict__ f,
                                                 __bf16* __restrict__ fb, int n) {
  int i = blockIdx.x * 256 + threadIdx.x;
  if (i < n) fb[i] = (__bf16)f[i];
}

// --- 2. weights f32 -> bf16 in WMMA B-fragment lane order; zero BN sums ----
// B fragment for V_WMMA_F32_16X16X32_BF16: per (k, ntile):
//   lane L<16 : column n = ntile*16+L, elements e=0..15 <-> K {0..7,16..23}
//   lane L>=16: column n = ntile*16+(L-16), elements e <-> K {8..15,24..31}
__global__ __launch_bounds__(256) void pack_w(const float* __restrict__ w,
                                              __bf16* __restrict__ wpk,
                                              float* __restrict__ sums) {
  int i = blockIdx.x * 256 + threadIdx.x;           // [0, 55296)
  if (i < 128) sums[i] = 0.0f;                      // zero sum[64] + sumsq[64]
  if (i >= WPK_ELEMS) return;
  int e    = i & 15;
  int lane = (i >> 4) & 31;
  int nt   = (i >> 9) & 3;
  int k    = i >> 11;
  int c    = ((e >> 3) << 4) + ((lane >> 4) << 3) + (e & 7); // K index 0..31
  int n    = nt * 16 + (lane & 15);                 // output channel
  wpk[i] = (__bf16)w[(k * CIN + c) * COUT + n];
}

// A fragment (16x32 bf16): lane<16 holds K {0..7,16..23}, lane>=16 holds
// K {8..15,24..31} of its row; zero fragment for missing neighbors.
__device__ __forceinline__ void load_a(Frag16& a, int idx,
                                       const __bf16* __restrict__ fb, int hi) {
  if (idx >= 0) {
    const __bf16* pf = fb + (size_t)idx * CIN;
    a.q[0] = *(const uint4*)(pf + 8 * hi);
    a.q[1] = *(const uint4*)(pf + 16 + 8 * hi);
  } else {
    a.q[0] = make_uint4(0u, 0u, 0u, 0u);
    a.q[1] = make_uint4(0u, 0u, 0u, 0u);
  }
}

// --- 3. gather + GEMM via v_wmma_f32_16x16x32_bf16 -------------------------
__global__ __launch_bounds__(256) void spconv_gemm(
    const __bf16* __restrict__ fb, const __bf16* __restrict__ wpk,
    const int* __restrict__ nbr, float* __restrict__ out,
    float* __restrict__ sum, float* __restrict__ sumsq) {
  __shared__ char smem[WPK_BYTES];

  // ---- stage packed B into LDS with CDNA5 async copies (ASYNCcnt) ----
  {
    // low 32 bits of the flat address of an LDS object == LDS byte offset
    unsigned lds_base = (unsigned)(uintptr_t)(void*)&smem[0];
    uint64_t gbase = (uint64_t)(const char*)wpk;
    unsigned off = threadIdx.x * 16u;
#pragma unroll 1
    for (int it = 0; it < WPK_BYTES / (256 * 16); ++it) {   // 27 iterations
      unsigned laddr = lds_base + off;
      uint64_t gaddr = gbase + off;
      // %2 (smem pointer) is unused in the template but forces the shared
      // array to escape, so the "memory" clobber covers the LDS writes.
      asm volatile("global_load_async_to_lds_b128 %0, %1, off"
                   :: "v"(laddr), "v"(gaddr), "v"((const void*)smem)
                   : "memory");
      off += 256u * 16u;
    }
    asm volatile("s_wait_asynccnt 0x0" ::: "memory");
  }
  __syncthreads();

  int wave = threadIdx.x >> 5;
  int lane = threadIdx.x & 31;
  int tile = blockIdx.x * WPB + wave;
  if (tile >= NTILES) return;                        // wave-uniform exit
  int hi   = lane >> 4;                              // lane half (A/B K split)
  int mrow = lane & 15;
  int r0   = tile * MT;
  int row  = r0 + mrow;

  v8f acc[4] = {};                                   // 16x64 f32 accumulators

  int idx_cur = nbr[row];                            // rulebook, k = 0
  Frag16 a_cur;
  load_a(a_cur, idx_cur, fb, hi);

#pragma unroll 3
  for (int k = 0; k < KOFF; ++k) {
    // software pipeline: fetch next index + gather before this offset's WMMAs
    int idx_nxt = (k + 1 < KOFF) ? nbr[(k + 1) * NV + row] : -1;
    Frag16 a_nxt;
    load_a(a_nxt, idx_nxt, fb, hi);

    const char* bbase = smem + (size_t)k * 4096 + (size_t)lane * 32;
#pragma unroll
    for (int nt = 0; nt < 4; ++nt) {
      Frag16 b;                                      // ds_load_b128 x2
      b.q[0] = *(const uint4*)(bbase + (size_t)nt * 1024);
      b.q[1] = *(const uint4*)(bbase + (size_t)nt * 1024 + 16);
      acc[nt] = __builtin_amdgcn_wmma_f32_16x16x32_bf16(
          /*neg_a=*/false, a_cur.v, /*neg_b=*/false, b.v,
          /*c_mod=*/(short)0, acc[nt], /*reuse_a=*/false, /*reuse_b=*/false);
    }
    a_cur = a_nxt;
  }

  // Epilogue: store f32 tile + per-channel partial sums for BN.
  // C/D layout: VGPR v, lanes 0-15: (M=v, N=lane); lanes 16-31: (M=8+v, N=lane-16)
#pragma unroll
  for (int nt = 0; nt < 4; ++nt) {
    int n = nt * 16 + mrow;
    float s = 0.0f, sq = 0.0f;
#pragma unroll
    for (int v = 0; v < 8; ++v) {
      float x = acc[nt][v];
      out[(size_t)(r0 + hi * 8 + v) * COUT + n] = x;
      s  += x;
      sq += x * x;
    }
    s  += __shfl_xor(s, 16, 32);                     // combine the two lane halves
    sq += __shfl_xor(sq, 16, 32);
    if (hi == 0) {
      atomicAdd(&sum[n], s);
      atomicAdd(&sumsq[n], sq);
    }
  }
}

// --- 4. BN statistics -> fused scale/shift ---------------------------------
__global__ void bn_finalize(const float* __restrict__ sum,
                            const float* __restrict__ sumsq,
                            const float* __restrict__ gamma,
                            const float* __restrict__ beta,
                            float* __restrict__ scale,
                            float* __restrict__ shift) {
  int c = threadIdx.x;
  if (c < COUT) {
    const float invN = 1.0f / (float)NV;
    float m   = sum[c] * invN;
    float var = sumsq[c] * invN - m * m;
    float inv = rsqrtf(var + 1e-5f);
    float sc  = inv * gamma[c];
    scale[c] = sc;
    shift[c] = beta[c] - m * sc;
  }
}

// --- 5. normalize + ReLU (in place on d_out) -------------------------------
__global__ __launch_bounds__(256) void bn_relu(float* __restrict__ out,
                                               const float* __restrict__ scale,
                                               const float* __restrict__ shift) {
  int i = blockIdx.x * 256 + threadIdx.x;
  if (i < NV * COUT) {
    int c = i & (COUT - 1);
    float x = out[i] * scale[c] + shift[c];
    out[i] = fmaxf(x, 0.0f);
  }
}

extern "C" void kernel_launch(void* const* d_in, const int* in_sizes, int n_in,
                              void* d_out, int out_size, void* d_ws, size_t ws_size,
                              hipStream_t stream) {
  const float* features = (const float*)d_in[0];   // [NV, CIN] f32
  const float* weight   = (const float*)d_in[1];   // [27, CIN, COUT] f32
  const float* gamma    = (const float*)d_in[2];   // [COUT] f32
  const float* beta     = (const float*)d_in[3];   // [COUT] f32
  const int*   nbr      = (const int*)d_in[4];     // [27, NV] i32
  float* out = (float*)d_out;                      // [NV, COUT] f32

  char* ws = (char*)d_ws;
  __bf16* fb    = (__bf16*)(ws + FB_OFF);
  __bf16* wpk   = (__bf16*)(ws + WPK_OFF);
  float*  sum   = (float*)(ws + SUM_OFF);
  float*  sumsq = (float*)(ws + SQ_OFF);
  float*  scale = (float*)(ws + SCALE_OFF);
  float*  shift = (float*)(ws + SHIFT_OFF);

  cvt_feats<<<(NV * CIN + 255) / 256, 256, 0, stream>>>(features, fb, NV * CIN);
  pack_w<<<(WPK_ELEMS + 255) / 256, 256, 0, stream>>>(weight, wpk, sum);
  spconv_gemm<<<(NTILES + WPB - 1) / WPB, 32 * WPB, 0, stream>>>(fb, wpk, nbr,
                                                                 out, sum, sumsq);
  bn_finalize<<<1, 64, 0, stream>>>(sum, sumsq, gamma, beta, scale, shift);
  bn_relu<<<(NV * COUT + 255) / 256, 256, 0, stream>>>(out, scale, shift);
}
